// BaseAttention_75917841924154
// MI455X (gfx1250) — matrix operand order repaired and checked by reading, not compiled, not run on hardware
//
#include <hip/hip_runtime.h>
#include <hip/hip_bf16.h>
#include <cmath>

// ---------------------------------------------------------------------------
// Types
// ---------------------------------------------------------------------------
typedef __attribute__((ext_vector_type(16))) __bf16        v16bf;
typedef __attribute__((ext_vector_type(8)))  float         v8f;
typedef __attribute__((ext_vector_type(4)))  unsigned int  u32x4;
typedef __attribute__((ext_vector_type(4)))  unsigned short u16x4;

union Frag {               // one WMMA 16-bit A/B operand (32 bytes, 8 VGPRs)
    v16bf          v;
    u32x4          u[2];
    unsigned short s[16];
};

union Pack8 {              // 8 bf16 -> one b128 store
    u32x4          u;
    unsigned short s[8];
};

__device__ __forceinline__ unsigned short f2bf(float f) {
    unsigned int u = __float_as_uint(f);
    u += 0x7FFFu + ((u >> 16) & 1u);      // round-to-nearest-even
    return (unsigned short)(u >> 16);
}

__device__ __forceinline__ v8f vzero() {
    v8f z = {0.f, 0.f, 0.f, 0.f, 0.f, 0.f, 0.f, 0.f};
    return z;
}

__device__ __forceinline__ v8f wmma_bf16(const Frag& a, const Frag& b, v8f c) {
    // D = A(16x32 bf16) * B(32x16 bf16) + C(16x16 f32)
    return __builtin_amdgcn_wmma_f32_16x16x32_bf16(false, a.v, false, b.v,
                                                   (short)0, c, false, false);
}

// lane <-> lane^16 exchange (wave32 half swap) without LDS.
__device__ __forceinline__ float xhalf16(float x) {
#if __has_builtin(__builtin_amdgcn_permlanex16)
    int r = __builtin_amdgcn_permlanex16(__float_as_int(x), __float_as_int(x),
                                         0x76543210, 0xfedcba98, false, false);
    return __int_as_float(r);
#else
    return __shfl_xor(x, 16, 32);
#endif
}

// Problem constants (fixed by setup_inputs)
#define BATCH  4
#define SEQ    2048
#define CHN    1024          // model dim = QK = V = O
#define HEADS  16
#define HDIM   64
#define MROWS  (BATCH * SEQ) // 8192

// ---------------------------------------------------------------------------
// Weight transpose + cast:  Wt[n,k] = bf16(W[k,n]),  K = N = 1024
// ---------------------------------------------------------------------------
__global__ __launch_bounds__(256) void transpose_w_kernel(
        const float* __restrict__ W, unsigned short* __restrict__ Wt) {
    __shared__ float tile[32][33];
    const int n0 = blockIdx.x * 32;
    const int k0 = blockIdx.y * 32;
    const int tx = threadIdx.x & 31;
    const int ty = threadIdx.x >> 5;            // 0..7
    #pragma unroll
    for (int i = 0; i < 4; ++i)
        tile[ty + i * 8][tx] = W[(size_t)(k0 + ty + i * 8) * CHN + n0 + tx];
    __syncthreads();
    #pragma unroll
    for (int i = 0; i < 4; ++i)
        Wt[(size_t)(n0 + ty + i * 8) * CHN + k0 + tx] = f2bf(tile[tx][ty + i * 8]);
}

// ---------------------------------------------------------------------------
// LayerNorm (row length 1024) + cast to bf16.  One block (256 thr) per row.
// ---------------------------------------------------------------------------
__global__ __launch_bounds__(256) void ln_bf16_kernel(
        const float* __restrict__ x, const float* __restrict__ w,
        const float* __restrict__ bvec, unsigned short* __restrict__ out) {
    const int row = blockIdx.x;
    const int t   = threadIdx.x;
    const float* xr = x + (size_t)row * CHN;

    float4 xv = *(const float4*)(xr + t * 4);
    float s  = xv.x + xv.y + xv.z + xv.w;
    float s2 = xv.x * xv.x + xv.y * xv.y + xv.z * xv.z + xv.w * xv.w;
    #pragma unroll
    for (int m = 16; m >= 1; m >>= 1) {
        s  += __shfl_xor(s,  m, 32);
        s2 += __shfl_xor(s2, m, 32);
    }
    __shared__ float ps[8], ps2[8];
    const int wid = t >> 5;
    if ((t & 31) == 0) { ps[wid] = s; ps2[wid] = s2; }
    __syncthreads();
    if (t == 0) {
        float a = 0.f, b2 = 0.f;
        #pragma unroll
        for (int i = 0; i < 8; ++i) { a += ps[i]; b2 += ps2[i]; }
        ps[0] = a; ps2[0] = b2;
    }
    __syncthreads();
    const float mu  = ps[0] * (1.0f / CHN);
    const float var = ps2[0] * (1.0f / CHN) - mu * mu;
    const float rs  = rsqrtf(var + 1e-5f);

    float4 wv = *(const float4*)(w + t * 4);
    float4 bv = *(const float4*)(bvec + t * 4);
    u16x4 r;
    r.x = f2bf((xv.x - mu) * rs * wv.x + bv.x);
    r.y = f2bf((xv.y - mu) * rs * wv.y + bv.y);
    r.z = f2bf((xv.z - mu) * rs * wv.z + bv.z);
    r.w = f2bf((xv.w - mu) * rs * wv.w + bv.w);
    *(u16x4*)(out + (size_t)row * CHN + t * 4) = r;
}

// ---------------------------------------------------------------------------
// bf16 WMMA GEMM:  out = alpha * (A[M,K] @ Wt[N,K]^T + bias)
//   A row-major bf16, Wt row-major bf16 (pre-transposed weight).
//   M = 8192, N = K = 1024.  Block: 256 thr = 8 waves, tile 128(M) x 256(N),
//   wave tile 64x64 (16 accumulators), K step 32.
//   OMODE 0: bf16 out[M,N]
//   OMODE 1: bf16 out written transposed-per-head: [(b,h,d), m]  (for V)
//   OMODE 2: f32  out[M,N]
// ---------------------------------------------------------------------------
template <int OMODE>
__global__ __launch_bounds__(256) void gemm_bf16_kernel(
        const unsigned short* __restrict__ A,
        const unsigned short* __restrict__ Wt,
        const float* __restrict__ bias,
        void* __restrict__ outv, float alpha) {
    const int wid  = threadIdx.x >> 5;
    const int lane = threadIdx.x & 31;
    const int n16  = lane & 15;
    const int hL   = lane >> 4;
    const int row0 = blockIdx.x * 128 + (wid >> 2) * 64;
    const int col0 = blockIdx.y * 256 + (wid & 3) * 64;

    v8f acc[4][4];
    #pragma unroll
    for (int i = 0; i < 4; ++i)
        #pragma unroll
        for (int j = 0; j < 4; ++j) acc[i][j] = vzero();

    for (int kk = 0; kk < CHN; kk += 32) {
        Frag af[4], bf[4];
        #pragma unroll
        for (int i = 0; i < 4; ++i) {           // A frag: 16 contiguous K elems @ hL*16
            const unsigned short* pa =
                A + (size_t)(row0 + i * 16 + n16) * CHN + kk + hL * 16;
            af[i].u[0] = *(const u32x4*)pa;
            af[i].u[1] = *(const u32x4*)(pa + 8);
        }
        #pragma unroll
        for (int j = 0; j < 4; ++j) {           // B frag: 16 contiguous K elems @ hL*16
            const unsigned short* pb =
                Wt + (size_t)(col0 + j * 16 + n16) * CHN + kk + hL * 16;
            bf[j].u[0] = *(const u32x4*)pb;
            bf[j].u[1] = *(const u32x4*)(pb + 8);
        }
        #pragma unroll
        for (int i = 0; i < 4; ++i)
            #pragma unroll
            for (int j = 0; j < 4; ++j)
                acc[i][j] = wmma_bf16(af[i], bf[j], acc[i][j]);
    }

    #pragma unroll
    for (int j = 0; j < 4; ++j) {
        const int col = col0 + j * 16 + n16;
        const float bb = bias[col];
        #pragma unroll
        for (int i = 0; i < 4; ++i) {
            if (OMODE == 1) {
                // mm is contiguous in r -> pack 8 bf16 into one b128 store
                const int head  = col >> 6, d = col & 63;
                const int rbase = row0 + i * 16 + 8 * hL;
                const int b = rbase >> 11, mm = rbase & 2047;
                Pack8 pk;
                #pragma unroll
                for (int r = 0; r < 8; ++r)
                    pk.s[r] = f2bf(alpha * (acc[i][j][r] + bb));
                *(u32x4*)((unsigned short*)outv +
                          (((size_t)((b * HEADS + head) * HDIM + d)) << 11) + mm) = pk.u;
            } else {
                #pragma unroll
                for (int r = 0; r < 8; ++r) {
                    const int rowg = row0 + i * 16 + r + 8 * hL;
                    const float val = alpha * (acc[i][j][r] + bb);
                    if (OMODE == 0)
                        ((unsigned short*)outv)[(size_t)rowg * CHN + col] = f2bf(val);
                    else
                        ((float*)outv)[(size_t)rowg * CHN + col] = val;
                }
            }
        }
    }
}

// ---------------------------------------------------------------------------
// Flash attention (bf16 WMMA, f32 softmax).
//   Q,K bf16 [b, n, 1024] (scale 1/8 folded into Q).  Vt bf16 [b, h, d, m].
//   Computes S^T = K·Q^T so each lane owns one query column: softmax stats
//   need only one lane^16 exchange (v_permlanex16).  The P^T operand of
//   O^T = V^T·P^T uses a permuted K-order (legal: WMMA sums over K) chosen so
//   every lane's B-fragment is exactly the 16 p-values it already holds --
//   zero cross-lane traffic; V^T fragment loads use the same permuted order
//   (lo @ kb+8*hL, hi @ +16), still contiguous 16B.
//   One wave = 16 queries; 4 waves/block; keys streamed in chunks of 32.
// ---------------------------------------------------------------------------
__global__ __launch_bounds__(128) void attn_kernel(
        const unsigned short* __restrict__ Q,
        const unsigned short* __restrict__ Kb,
        const unsigned short* __restrict__ Vt,
        unsigned short* __restrict__ O) {
    const int b    = blockIdx.z;
    const int h    = blockIdx.y;
    const int wid  = threadIdx.x >> 5;
    const int lane = threadIdx.x & 31;
    const int n16  = lane & 15;           // query column within tile
    const int hL   = lane >> 4;
    const int qb   = blockIdx.x * 64 + wid * 16;

    // Hoisted Q B-fragments (channels 0-31 and 32-63 of this head)
    Frag qf[2];
    {
        const unsigned short* pq =
            Q + (size_t)(b * SEQ + qb + n16) * CHN + h * HDIM + hL * 16;
        qf[0].u[0] = *(const u32x4*)pq;
        qf[0].u[1] = *(const u32x4*)(pq + 8);
        qf[1].u[0] = *(const u32x4*)(pq + 32);
        qf[1].u[1] = *(const u32x4*)(pq + 40);
    }

    v8f o[4];
    #pragma unroll
    for (int t = 0; t < 4; ++t) o[t] = vzero();
    float Mrun = -INFINITY, lrun = 0.f;

    for (int kb = 0; kb < SEQ; kb += 32) {
        // S^T chunk: two 16-key tiles, K-dim = 64 channels (2 WMMA each)
        v8f s[2];
        #pragma unroll
        for (int kt = 0; kt < 2; ++kt) {
            Frag kf0, kf1;
            const unsigned short* pk =
                Kb + (size_t)(b * SEQ + kb + kt * 16 + n16) * CHN + h * HDIM + hL * 16;
            kf0.u[0] = *(const u32x4*)pk;
            kf0.u[1] = *(const u32x4*)(pk + 8);
            kf1.u[0] = *(const u32x4*)(pk + 32);
            kf1.u[1] = *(const u32x4*)(pk + 40);
            v8f a = vzero();
            a = wmma_bf16(kf0, qf[0], a);
            a = wmma_bf16(kf1, qf[1], a);
            s[kt] = a;
        }

        // Online softmax (per lane = per query; lane pair L<->L+16 shares stats)
        float cmax = -INFINITY;
        #pragma unroll
        for (int r = 0; r < 8; ++r)
            cmax = fmaxf(cmax, fmaxf(s[0][r], s[1][r]));
        cmax = fmaxf(cmax, xhalf16(cmax));
        const float Mnew = fmaxf(Mrun, cmax);
        const float corr = __expf(Mrun - Mnew);
        float p0[8], p1[8], csum = 0.f;
        #pragma unroll
        for (int r = 0; r < 8; ++r) {
            p0[r] = __expf(s[0][r] - Mnew);
            p1[r] = __expf(s[1][r] - Mnew);
            csum += p0[r] + p1[r];
        }
        csum += xhalf16(csum);
        lrun = lrun * corr + csum;
        Mrun = Mnew;
        #pragma unroll
        for (int t = 0; t < 4; ++t)
            #pragma unroll
            for (int r = 0; r < 8; ++r) o[t][r] *= corr;

        // P^T B-fragment in permuted K-order: lane-local, no shuffles.
        Frag pf;
        #pragma unroll
        for (int r = 0; r < 8; ++r) {
            pf.s[r]     = f2bf(p0[r]);   // keys kb + 8*hL + 0..7
            pf.s[8 + r] = f2bf(p1[r]);   // keys kb + 16 + 8*hL + 0..7
        }

        // O^T += V^T(16d x 32keys) · P^T(32keys x 16q), same permuted K-order
        #pragma unroll
        for (int dt = 0; dt < 4; ++dt) {
            Frag vf;
            const unsigned short* pv =
                Vt + (size_t)((b * HEADS + h) * HDIM + dt * 16 + n16) * SEQ + kb + hL * 8;
            vf.u[0] = *(const u32x4*)pv;
            vf.u[1] = *(const u32x4*)(pv + 16);
            o[dt] = wmma_bf16(vf, pf, o[dt]);
        }
    }

    // Epilogue: d-index contiguous in r -> pack 8 bf16, one b128 store per tile
    const float inv = 1.0f / lrun;
    unsigned short* ob =
        O + (size_t)(b * SEQ + qb + n16) * CHN + h * HDIM + 8 * hL;
    #pragma unroll
    for (int dt = 0; dt < 4; ++dt) {
        Pack8 pk;
        #pragma unroll
        for (int r = 0; r < 8; ++r) pk.s[r] = f2bf(o[dt][r] * inv);
        *(u32x4*)(ob + dt * 16) = pk.u;
    }
}

// ---------------------------------------------------------------------------
// Launcher
// ---------------------------------------------------------------------------
extern "C" void kernel_launch(void* const* d_in, const int* in_sizes, int n_in,
                              void* d_out, int out_size, void* d_ws, size_t ws_size,
                              hipStream_t stream) {
    const float* inputs_q  = (const float*)d_in[0];
    const float* inputs_kv = (const float*)d_in[1];
    const float* ln_q_w = (const float*)d_in[2];
    const float* ln_q_b = (const float*)d_in[3];
    const float* ln_k_w = (const float*)d_in[4];
    const float* ln_k_b = (const float*)d_in[5];
    const float* ln_v_w = (const float*)d_in[6];
    const float* ln_v_b = (const float*)d_in[7];
    const float* Wq = (const float*)d_in[8];
    const float* bq = (const float*)d_in[9];
    const float* Wk = (const float*)d_in[10];
    const float* bk = (const float*)d_in[11];
    const float* Wv = (const float*)d_in[12];
    const float* bv = (const float*)d_in[13];
    const float* Wo = (const float*)d_in[14];
    const float* bo = (const float*)d_in[15];

    // Workspace carve-up (bf16 buffers)
    const size_t ACT = (size_t)MROWS * CHN;   // 8M elems
    const size_t WEL = (size_t)CHN * CHN;     // 1M elems
    char* ws = (char*)d_ws;
    size_t off = 0;
    auto carve = [&](size_t bytes) { char* p = ws + off; off += (bytes + 255) & ~(size_t)255; return p; };
    unsigned short* lnq  = (unsigned short*)carve(ACT * 2);
    unsigned short* lnk  = (unsigned short*)carve(ACT * 2);
    unsigned short* lnv  = (unsigned short*)carve(ACT * 2);
    unsigned short* qbuf = (unsigned short*)carve(ACT * 2);
    unsigned short* kbuf = (unsigned short*)carve(ACT * 2);
    unsigned short* vtb  = (unsigned short*)carve(ACT * 2);
    unsigned short* attn = (unsigned short*)carve(ACT * 2);
    unsigned short* Wqt  = (unsigned short*)carve(WEL * 2);
    unsigned short* Wkt  = (unsigned short*)carve(WEL * 2);
    unsigned short* Wvt  = (unsigned short*)carve(WEL * 2);
    unsigned short* Wot  = (unsigned short*)carve(WEL * 2);

    // 1) Weight transposes (fp32 -> bf16, [K,N] -> [N,K])
    dim3 tgrid(CHN / 32, CHN / 32);
    transpose_w_kernel<<<tgrid, 256, 0, stream>>>(Wq, Wqt);
    transpose_w_kernel<<<tgrid, 256, 0, stream>>>(Wk, Wkt);
    transpose_w_kernel<<<tgrid, 256, 0, stream>>>(Wv, Wvt);
    transpose_w_kernel<<<tgrid, 256, 0, stream>>>(Wo, Wot);

    // 2) LayerNorms
    ln_bf16_kernel<<<MROWS, 256, 0, stream>>>(inputs_q,  ln_q_w, ln_q_b, lnq);
    ln_bf16_kernel<<<MROWS, 256, 0, stream>>>(inputs_kv, ln_k_w, ln_k_b, lnk);
    ln_bf16_kernel<<<MROWS, 256, 0, stream>>>(inputs_kv, ln_v_w, ln_v_b, lnv);

    // 3) Projections.  1/sqrt(head_dim) folded into Q.
    dim3 ggrid(MROWS / 128, CHN / 256);
    gemm_bf16_kernel<0><<<ggrid, 256, 0, stream>>>(lnq, Wqt, bq, qbuf, 0.125f);
    gemm_bf16_kernel<0><<<ggrid, 256, 0, stream>>>(lnk, Wkt, bk, kbuf, 1.0f);
    gemm_bf16_kernel<1><<<ggrid, 256, 0, stream>>>(lnv, Wvt, bv, vtb,  1.0f);

    // 4) Attention
    dim3 agrid(SEQ / 64, HEADS, BATCH);
    attn_kernel<<<agrid, 128, 0, stream>>>(qbuf, kbuf, vtb, attn);

    // 5) Output projection (fp32 out)
    gemm_bf16_kernel<2><<<ggrid, 256, 0, stream>>>(attn, Wot, bo, (float*)d_out, 1.0f);
}
